// TransformerEncoder_58248346468663
// MI455X (gfx1250) — matrix operand order repaired and checked by reading, not compiled, not run on hardware
//
#include <hip/hip_runtime.h>
#include <hip/hip_bf16.h>
#include <cstddef>

typedef unsigned short u16;
typedef __attribute__((ext_vector_type(16))) __bf16 v16bf;
typedef __attribute__((ext_vector_type(8)))  float  v8f;

#define B_   32
#define S_   512
#define D_   512
#define H_   8
#define DK_  64
#define DFF_ 2048
#define L_   4
#define BS_  (B_ * S_)          // 16384
#define SD_  (S_ * D_)          // 262144
#define BSD_ ((size_t)BS_ * D_) // 8388608
#define COUT_ 100

// ---------------- helpers ----------------

__device__ __forceinline__ u16 f2bf(float f) {
  union { float f; unsigned u; } x; x.f = f;
  unsigned r = x.u + 0x7FFFu + ((x.u >> 16) & 1u);   // round-to-nearest-even
  return (u16)(r >> 16);
}

union Frag16 { uint4 q[2]; v16bf f; };

__device__ __forceinline__ v8f vzero8() {
  v8f z;
#pragma unroll
  for (int i = 0; i < 8; ++i) z[i] = 0.0f;
  return z;
}

__device__ __forceinline__ v8f wmma_bf16(v16bf a, v16bf b, v8f c) {
  return __builtin_amdgcn_wmma_f32_16x16x32_bf16(false, a, false, b, (short)0, c,
                                                 false, false);
}

// A fragment: 16x32 bf16, A row-major [lda], tile origin (row0,k0).
// lane l: m = l%16 ; K runs: (l/16)*8 + [0..7] and +16..23  (ISA 16-bit A layout)
__device__ __forceinline__ v16bf load_frag_a(const u16* A, int lda, int row0, int k0) {
  int lane = threadIdx.x & 31;
  const u16* p = A + (size_t)(row0 + (lane & 15)) * lda + k0 + ((lane >> 4) << 3);
  Frag16 t;
  t.q[0] = *(const uint4*)(p);
  t.q[1] = *(const uint4*)(p + 16);
  return t.f;
}

// B fragment: 32x16 bf16 for D=A*B+C, B supplied transposed [N,K] row-major.
// lane l: n = n0 + l%16 ; K runs: (l/16)*8 + [0..7] and +16..23  (ISA 16-bit B layout)
__device__ __forceinline__ v16bf load_frag_b(const u16* Bt, int ldb, int n0, int k0) {
  int lane = threadIdx.x & 31;
  const u16* p = Bt + (size_t)(n0 + (lane & 15)) * ldb + k0 + ((lane >> 4) << 3);
  Frag16 t;
  t.q[0] = *(const uint4*)(p);
  t.q[1] = *(const uint4*)(p + 16);
  return t.f;
}

// A fragment from a 16x32 row-major bf16 tile in LDS
__device__ __forceinline__ v16bf load_frag_a_lds(const u16* p0) {
  int lane = threadIdx.x & 31;
  const u16* p = p0 + (lane & 15) * 32 + ((lane >> 4) << 3);
  Frag16 t;
  t.q[0] = *(const uint4*)(p);
  t.q[1] = *(const uint4*)(p + 16);
  return t.f;
}

// ---------------- weight transpose + bf16 convert: in[K,N] f32 -> out[N,K] bf16 ----

__global__ __launch_bounds__(256) void transpose_f32_bf16(const float* __restrict__ in,
                                                          u16* __restrict__ out,
                                                          int K, int N) {
  __shared__ float tile[32][33];
  int kb = blockIdx.x * 32, nb = blockIdx.y * 32;
  int tx = threadIdx.x & 31, ty = threadIdx.x >> 5;   // 8 rows of 32
#pragma unroll
  for (int r = ty; r < 32; r += 8)
    tile[r][tx] = in[(size_t)(kb + r) * N + nb + tx];
  __syncthreads();
#pragma unroll
  for (int r = ty; r < 32; r += 8)
    out[(size_t)(nb + r) * K + kb + tx] = f2bf(tile[tx][r]);
}

// ---------------- input projection: u = einsum(bcs,cd) + Wp_b + W_pos ------------

__global__ __launch_bounds__(256) void proj_kernel(const float* __restrict__ x,
                                                   const float* __restrict__ wp,
                                                   const float* __restrict__ wpb,
                                                   const float* __restrict__ wpos,
                                                   float* __restrict__ hf,
                                                   u16* __restrict__ hb) {
  int s = blockIdx.x, b = blockIdx.y;
  for (int d = threadIdx.x; d < D_; d += 256) {
    float acc = wpb[d] + wpos[(size_t)s * D_ + d];
#pragma unroll
    for (int c = 0; c < 6; ++c)
      acc += x[((size_t)b * 6 + c) * S_ + s] * wp[c * D_ + d];
    size_t i = ((size_t)b * S_ + s) * D_ + d;
    hf[i] = acc;
    hb[i] = f2bf(acc);
  }
}

// ---------------- generic bf16 WMMA GEMM: C[M,N] = A[M,K] * Bt[N,K]^T -----------
// block tile 128(M) x 256(N), 8 waves as 2(M) x 4(N); wave tile 64x64 =
// 16 WMMAs per k-step against 16 b128 loads (wmma:vmem = 1.0).
// epilogue: +bias[N], gelu, +resid[M,N]; writes outF (f32) and/or outB (bf16)

__global__ __launch_bounds__(256) void gemm_bf16(const u16* __restrict__ A,
                                                 const u16* __restrict__ Bt,
                                                 const float* __restrict__ bias,
                                                 const float* __restrict__ resid,
                                                 float* __restrict__ outF,
                                                 u16* __restrict__ outB,
                                                 int M, int N, int K, int gelu) {
  int wid = threadIdx.x >> 5, lane = threadIdx.x & 31;
  int wm = wid & 1, wn = wid >> 1;                 // 2 x 4 wave grid
  int m0 = blockIdx.y * 128 + wm * 64;             // wave: 64 rows
  int n0 = blockIdx.x * 256 + wn * 64;             // wave: 64 cols
  v8f acc[4][4];
#pragma unroll
  for (int i = 0; i < 4; ++i)
#pragma unroll
    for (int j = 0; j < 4; ++j) acc[i][j] = vzero8();

  int la = lane & 15, lh8 = (lane >> 4) << 3, lh16 = (lane >> 4) << 4;

  for (int k = 0; k < K; k += 32) {
    // prefetch the k+64 tiles into cache while computing (global_prefetch_b8)
    if (k + 64 < K) {
      __builtin_prefetch(A + (size_t)(m0 + la) * K + (k + 64) + lh8, 0, 1);
      __builtin_prefetch(Bt + (size_t)(n0 + la) * K + (k + 64) + lh16, 0, 1);
      __builtin_prefetch(A + (size_t)(m0 + 32 + la) * K + (k + 64) + lh8, 0, 1);
      __builtin_prefetch(Bt + (size_t)(n0 + 32 + la) * K + (k + 64) + lh16, 0, 1);
    }
    v16bf af[4];
#pragma unroll
    for (int i = 0; i < 4; ++i) af[i] = load_frag_a(A, K, m0 + 16 * i, k);
#pragma unroll
    for (int j = 0; j < 4; ++j) {
      v16bf bf = load_frag_b(Bt, K, n0 + 16 * j, k);
#pragma unroll
      for (int i = 0; i < 4; ++i) acc[i][j] = wmma_bf16(af[i], bf, acc[i][j]);
    }
  }

  int mh = lh8;               // +8 rows for lanes 16..31
  int nl = la;
#pragma unroll
  for (int i = 0; i < 4; ++i)
#pragma unroll
    for (int j = 0; j < 4; ++j)
#pragma unroll
      for (int r = 0; r < 8; ++r) {
        int m = m0 + i * 16 + r + mh;
        int n = n0 + j * 16 + nl;
        float v = acc[i][j][r];
        if (bias) v += bias[n];
        if (gelu) v = 0.5f * v * (1.0f + erff(v * 0.70710678118654752f));
        size_t idx = (size_t)m * N + n;
        if (resid) v += resid[idx];
        if (outF) outF[idx] = v;
        if (outB) outB[idx] = f2bf(v);
      }
}

// ---------------- V transpose: vb[b,s,h*64+d] (bf16) -> vt[(b*8+h)*64+d, s] ------

__global__ __launch_bounds__(256) void vt_kernel(const u16* __restrict__ vb,
                                                 u16* __restrict__ vt) {
  __shared__ u16 t[64][65];
  int bh = blockIdx.x;                 // b*8+h
  int b = bh >> 3, h = bh & 7;
  int s0 = blockIdx.y * 64;
  int c = threadIdx.x & 63, r = threadIdx.x >> 6;  // 4 rows of 64
#pragma unroll
  for (int sl = r; sl < 64; sl += 4)
    t[sl][c] = vb[((size_t)(b * S_ + s0 + sl)) * D_ + h * DK_ + c];
  __syncthreads();
#pragma unroll
  for (int d = r; d < 64; d += 4)
    vt[((size_t)(bh * DK_ + d)) * S_ + s0 + c] = t[c][d];
}

// ---------------- flash attention (bf16 WMMA, online softmax) --------------------
// Qb,Kb: [B,S,H,DK] bf16 ; Vt: [B*H, DK, S] bf16 ; Ob: [B,S,H,DK] bf16

__global__ __launch_bounds__(128) void attn_kernel(const u16* __restrict__ Qb,
                                                   const u16* __restrict__ Kb,
                                                   const u16* __restrict__ Vt,
                                                   u16* __restrict__ Ob) {
  int bh = blockIdx.x;
  int b = bh >> 3, h = bh & 7;
  int wid = threadIdx.x >> 5, lane = threadIdx.x & 31;
  int q0 = blockIdx.y * 64 + wid * 16;

  const u16* Qp = Qb + ((size_t)(b * S_ + q0) * D_ + h * DK_);
  const u16* Kp = Kb + ((size_t)b * S_ * D_ + h * DK_);
  const u16* Vp = Vt + (size_t)bh * DK_ * S_;

  v16bf qf0 = load_frag_a(Qp, D_, 0, 0);
  v16bf qf1 = load_frag_a(Qp, D_, 0, 32);

  v8f c0 = vzero8(), c1 = vzero8(), c2 = vzero8(), c3 = vzero8();
  float rmax[8], rsum[8];
#pragma unroll
  for (int r = 0; r < 8; ++r) { rmax[r] = -3.0e38f; rsum[r] = 0.0f; }

  __shared__ __align__(32) u16 pbuf[4][16 * 32];
  u16* myp = pbuf[wid];

  int mh = (lane >> 4) << 3;
  int nl = lane & 15;

  for (int st = 0; st < S_; st += 32) {
    const u16* Kt = Kp + (size_t)st * D_;
    v8f s0 = vzero8(), s1 = vzero8();
    {
      v16bf b00 = load_frag_b(Kt, D_, 0, 0);
      v16bf b01 = load_frag_b(Kt, D_, 0, 32);
      v16bf b10 = load_frag_b(Kt, D_, 16, 0);
      v16bf b11 = load_frag_b(Kt, D_, 16, 32);
      s0 = wmma_bf16(qf0, b00, s0);
      s0 = wmma_bf16(qf1, b01, s0);
      s1 = wmma_bf16(qf0, b10, s1);
      s1 = wmma_bf16(qf1, b11, s1);
    }
#pragma unroll
    for (int r = 0; r < 8; ++r) {
      float a = s0[r] * 0.125f;      // 1/sqrt(64)
      float bb = s1[r] * 0.125f;
      float mx = fmaxf(a, bb);
      mx = fmaxf(mx, __shfl_xor(mx, 1, 16));
      mx = fmaxf(mx, __shfl_xor(mx, 2, 16));
      mx = fmaxf(mx, __shfl_xor(mx, 4, 16));
      mx = fmaxf(mx, __shfl_xor(mx, 8, 16));
      float nm = fmaxf(rmax[r], mx);
      float cf = __expf(rmax[r] - nm);
      float e0 = __expf(a - nm);
      float e1 = __expf(bb - nm);
      float rs = e0 + e1;
      rs += __shfl_xor(rs, 1, 16);
      rs += __shfl_xor(rs, 2, 16);
      rs += __shfl_xor(rs, 4, 16);
      rs += __shfl_xor(rs, 8, 16);
      rsum[r] = rsum[r] * cf + rs;
      rmax[r] = nm;
      c0[r] *= cf; c1[r] *= cf; c2[r] *= cf; c3[r] *= cf;
      int m = r + mh;
      myp[m * 32 + nl]      = f2bf(e0);
      myp[m * 32 + 16 + nl] = f2bf(e1);
    }
    asm volatile("s_wait_dscnt 0" ::: "memory");
    v16bf pf = load_frag_a_lds(myp);
    v16bf v0 = load_frag_b(Vp, S_, 0,  st);
    v16bf v1 = load_frag_b(Vp, S_, 16, st);
    v16bf v2 = load_frag_b(Vp, S_, 32, st);
    v16bf v3 = load_frag_b(Vp, S_, 48, st);
    c0 = wmma_bf16(pf, v0, c0);
    c1 = wmma_bf16(pf, v1, c1);
    c2 = wmma_bf16(pf, v2, c2);
    c3 = wmma_bf16(pf, v3, c3);
  }

#pragma unroll
  for (int r = 0; r < 8; ++r) {
    float inv = 1.0f / rsum[r];
    int m = r + mh;
    size_t row = ((size_t)(b * S_ + q0 + m)) * D_ + h * DK_ + nl;
    Ob[row]      = f2bf(c0[r] * inv);
    Ob[row + 16] = f2bf(c1[r] * inv);
    Ob[row + 32] = f2bf(c2[r] * inv);
    Ob[row + 48] = f2bf(c3[r] * inv);
  }
}

// ---------------- batchnorm over (B,S) per channel ------------------------------

__global__ void zero_kernel(float* p, int n) {
  int i = blockIdx.x * blockDim.x + threadIdx.x;
  if (i < n) p[i] = 0.0f;
}

__global__ __launch_bounds__(512) void bn_partial(const float* __restrict__ x,
                                                  float* __restrict__ sum,
                                                  float* __restrict__ sq) {
  int d = threadIdx.x;
  int r0 = blockIdx.x * 32;
  float s = 0.0f, q = 0.0f;
#pragma unroll 4
  for (int r = 0; r < 32; ++r) {
    float v = x[((size_t)(r0 + r)) * D_ + d];
    s += v; q += v * v;
  }
  atomicAdd(&sum[d], s);
  atomicAdd(&sq[d], q);
}

__global__ __launch_bounds__(512) void bn_finalize(float* sum, float* sq) {
  int d = threadIdx.x;
  float m = sum[d] * (1.0f / (float)BS_);
  float v = fmaxf(sq[d] * (1.0f / (float)BS_) - m * m, 0.0f);
  sum[d] = m; sq[d] = v;
}

__global__ __launch_bounds__(256) void bn_apply(float* __restrict__ h,
                                                u16* __restrict__ hb,
                                                const float* __restrict__ mean,
                                                const float* __restrict__ var,
                                                const float* __restrict__ g,
                                                const float* __restrict__ be) {
  size_t i = (size_t)blockIdx.x * 256 + threadIdx.x;
  int d = (int)(i & (D_ - 1));
  float v = (h[i] - mean[d]) * rsqrtf(var[d] + 1e-5f) * g[d] + be[d];
  h[i] = v;
  hb[i] = f2bf(v);
}

// ---------------- head: out[b,j] = z[b,:] . head_w[:,j] + head_b[j] -------------

__global__ __launch_bounds__(256) void head_kernel(const float* __restrict__ z,
                                                   const float* __restrict__ w,
                                                   const float* __restrict__ hb,
                                                   float* __restrict__ out) {
  int j = blockIdx.x, b = blockIdx.y;
  const float* zb = z + (size_t)b * SD_;
  float acc = 0.0f;
  for (int k = threadIdx.x; k < SD_; k += 256)
    acc += zb[k] * w[(size_t)k * COUT_ + j];
  __shared__ float red[256];
  red[threadIdx.x] = acc;
  __syncthreads();
  for (int s = 128; s > 0; s >>= 1) {
    if (threadIdx.x < s) red[threadIdx.x] += red[threadIdx.x + s];
    __syncthreads();
  }
  if (threadIdx.x == 0) out[b * COUT_ + j] = red[0] + hb[j];
}

// ---------------- orchestration --------------------------------------------------

extern "C" void kernel_launch(void* const* d_in, const int* in_sizes, int n_in,
                              void* d_out, int out_size, void* d_ws, size_t ws_size,
                              hipStream_t stream) {
  (void)in_sizes; (void)n_in; (void)out_size;
  const float* x      = (const float*)d_in[0];
  const float* Wp_w   = (const float*)d_in[1];
  const float* Wp_b   = (const float*)d_in[2];
  const float* W_pos  = (const float*)d_in[3];
  const float* WQ     = (const float*)d_in[4];
  const float* WK     = (const float*)d_in[5];
  const float* WV     = (const float*)d_in[6];
  const float* WO     = (const float*)d_in[7];
  const float* f1w    = (const float*)d_in[8];
  const float* f1b    = (const float*)d_in[9];
  const float* f2w    = (const float*)d_in[10];
  const float* f2b    = (const float*)d_in[11];
  const float* bn1g   = (const float*)d_in[12];
  const float* bn1b   = (const float*)d_in[13];
  const float* bn2g   = (const float*)d_in[14];
  const float* bn2b   = (const float*)d_in[15];
  const float* head_w = (const float*)d_in[16];
  const float* head_b = (const float*)d_in[17];
  float* out = (float*)d_out;

  char* ws = (char*)d_ws;
  size_t off = 0;
  auto take = [&](size_t bytes) -> char* {
    char* p = ws + off;
    off += (bytes + 255) & ~(size_t)255;
    return p;
  };
  float* hf   = (float*)take(BSD_ * 4);
  u16*   hb   = (u16*)take(BSD_ * 2);
  u16*   qb   = (u16*)take(BSD_ * 2);
  u16*   kb2  = (u16*)take(BSD_ * 2);
  u16*   vb2  = (u16*)take(BSD_ * 2);
  u16*   vt   = (u16*)take(BSD_ * 2);
  u16*   cb   = (u16*)take(BSD_ * 2);
  u16*   f1o  = (u16*)take((size_t)BS_ * DFF_ * 2);
  const size_t LSTRIDE = (size_t)4 * D_ * D_ + 2 * (size_t)D_ * DFF_; // 3,145,728
  u16*   wts  = (u16*)take(LSTRIDE * L_ * 2);
  float* stats = (float*)take(2 * D_ * 4);
  if (off > ws_size) return;  // workspace too small; harness sizes generously

  // --- convert + transpose all weights to bf16 [N,K] ---
  for (int l = 0; l < L_; ++l) {
    u16* wqt = wts + (size_t)l * LSTRIDE;
    u16* wkt = wqt + (size_t)D_ * D_;
    u16* wvt = wkt + (size_t)D_ * D_;
    u16* wot = wvt + (size_t)D_ * D_;
    u16* f1t = wot + (size_t)D_ * D_;
    u16* f2t = f1t + (size_t)D_ * DFF_;
    transpose_f32_bf16<<<dim3(16, 16), 256, 0, stream>>>(WQ + (size_t)l * D_ * D_, wqt, D_, D_);
    transpose_f32_bf16<<<dim3(16, 16), 256, 0, stream>>>(WK + (size_t)l * D_ * D_, wkt, D_, D_);
    transpose_f32_bf16<<<dim3(16, 16), 256, 0, stream>>>(WV + (size_t)l * D_ * D_, wvt, D_, D_);
    transpose_f32_bf16<<<dim3(16, 16), 256, 0, stream>>>(WO + (size_t)l * D_ * D_, wot, D_, D_);
    transpose_f32_bf16<<<dim3(16, 64), 256, 0, stream>>>(f1w + (size_t)l * D_ * DFF_, f1t, D_, DFF_);
    transpose_f32_bf16<<<dim3(64, 16), 256, 0, stream>>>(f2w + (size_t)l * DFF_ * D_, f2t, DFF_, D_);
  }

  // --- input projection + positional encoding ---
  proj_kernel<<<dim3(S_, B_), 256, 0, stream>>>(x, Wp_w, Wp_b, W_pos, hf, hb);

  for (int l = 0; l < L_; ++l) {
    u16* wqt = wts + (size_t)l * LSTRIDE;
    u16* wkt = wqt + (size_t)D_ * D_;
    u16* wvt = wkt + (size_t)D_ * D_;
    u16* wot = wvt + (size_t)D_ * D_;
    u16* f1t = wot + (size_t)D_ * D_;
    u16* f2t = f1t + (size_t)D_ * DFF_;

    // Q, K, V projections (bf16 out); grid: (N/256, M/128)
    gemm_bf16<<<dim3(2, 128), 256, 0, stream>>>(hb, wqt, nullptr, nullptr, nullptr, qb,
                                                BS_, D_, D_, 0);
    gemm_bf16<<<dim3(2, 128), 256, 0, stream>>>(hb, wkt, nullptr, nullptr, nullptr, kb2,
                                                BS_, D_, D_, 0);
    gemm_bf16<<<dim3(2, 128), 256, 0, stream>>>(hb, wvt, nullptr, nullptr, nullptr, vb2,
                                                BS_, D_, D_, 0);
    // V -> V^T per (b,h)
    vt_kernel<<<dim3(B_ * H_, S_ / 64), 256, 0, stream>>>(vb2, vt);
    // flash attention
    attn_kernel<<<dim3(B_ * H_, S_ / 64), 128, 0, stream>>>(qb, kb2, vt, cb);
    // output projection + residual (f32 out)
    gemm_bf16<<<dim3(2, 128), 256, 0, stream>>>(cb, wot, nullptr, hf, hf, nullptr,
                                                BS_, D_, D_, 0);
    // BN1
    zero_kernel<<<1, 1024, 0, stream>>>(stats, 2 * D_);
    bn_partial<<<BS_ / 32, 512, 0, stream>>>(hf, stats, stats + D_);
    bn_finalize<<<1, 512, 0, stream>>>(stats, stats + D_);
    bn_apply<<<(unsigned)(BSD_ / 256), 256, 0, stream>>>(hf, hb, stats, stats + D_,
                                                         bn1g + l * D_, bn1b + l * D_);
    // FFN: gelu(h @ f1 + b1) @ f2 + b2 + residual
    gemm_bf16<<<dim3(8, 128), 256, 0, stream>>>(hb, f1t, f1b + (size_t)l * DFF_,
                                                 nullptr, nullptr, f1o,
                                                 BS_, DFF_, D_, 1);
    gemm_bf16<<<dim3(2, 128), 256, 0, stream>>>(f1o, f2t, f2b + (size_t)l * D_,
                                                hf, hf, nullptr,
                                                BS_, D_, DFF_, 0);
    // BN2
    zero_kernel<<<1, 1024, 0, stream>>>(stats, 2 * D_);
    bn_partial<<<BS_ / 32, 512, 0, stream>>>(hf, stats, stats + D_);
    bn_finalize<<<1, 512, 0, stream>>>(stats, stats + D_);
    bn_apply<<<(unsigned)(BSD_ / 256), 256, 0, stream>>>(hf, hb, stats, stats + D_,
                                                         bn2g + l * D_, bn2b + l * D_);
  }

  head_kernel<<<dim3(COUT_, B_), 256, 0, stream>>>(hf, head_w, head_b, out);
}
